// GatedDeltaRuleModel_10179072491845
// MI455X (gfx1250) — compile-verified, matrix operand order checked by reading, not run on hardware
//
#include <hip/hip_runtime.h>
#include <hip/hip_bf16.h>

// ---------------------------------------------------------------------------
// Types for CDNA5 WMMA
// ---------------------------------------------------------------------------
typedef __attribute__((ext_vector_type(16))) __bf16       bf16x16;
typedef __attribute__((ext_vector_type(8)))  float        v8f;
typedef __attribute__((ext_vector_type(8)))  unsigned int uint8v;

// Problem constants (match the reference)
#define HIDDEN 2048
#define HK 16
#define HV 32
#define DK 128
#define DV 128
#define BB 2
#define SS 2048
#define MROWS (BB * SS)          // 4096 token rows

// GEMM tiling
#define BM 128
#define BN 128
#define BKK 32
#define TILE_ELEMS (BM * BKK)            // bf16 elements per tile (A or B)
#define TILE_BYTES (TILE_ELEMS * 2)      // 8 KB

// float -> bf16, round-to-nearest-even
__device__ __forceinline__ unsigned short f2bf(float f) {
    unsigned int u = __builtin_bit_cast(unsigned int, f);
    unsigned int r = u + 0x7FFFu + ((u >> 16) & 1u);
    return (unsigned short)(r >> 16);
}

// ---------------------------------------------------------------------------
// 1) Embedding gather: x[row][:] = embed[ids[row]][:]  (+ optional bf16 copy)
// ---------------------------------------------------------------------------
__global__ __launch_bounds__(256) void embed_kernel(const int* __restrict__ ids,
                                                    const float* __restrict__ emb,
                                                    float* __restrict__ x,
                                                    unsigned short* __restrict__ xb) {
    int row = blockIdx.x;
    int id  = ids[row];
    const float4* src = (const float4*)(emb + (size_t)id * HIDDEN);
    float4* dst       = (float4*)(x + (size_t)row * HIDDEN);
    for (int i = threadIdx.x; i < HIDDEN / 4; i += 256) {
        float4 f = src[i];
        dst[i] = f;
        if (xb) {
            unsigned lo = (unsigned)f2bf(f.x) | ((unsigned)f2bf(f.y) << 16);
            unsigned hi = (unsigned)f2bf(f.z) | ((unsigned)f2bf(f.w) << 16);
            uint2 p; p.x = lo; p.y = hi;
            *(uint2*)(xb + (size_t)row * HIDDEN + i * 4) = p;
        }
    }
}

// ---------------------------------------------------------------------------
// 1b) f32 -> bf16 cast (weights, grid-stride)
// ---------------------------------------------------------------------------
__global__ __launch_bounds__(256) void cast_f32_bf16(const float* __restrict__ in,
                                                     unsigned short* __restrict__ out,
                                                     long long n) {
    long long i      = ((long long)blockIdx.x * 256 + threadIdx.x) * 4;
    long long stride = (long long)gridDim.x * 256 * 4;
    for (; i < n; i += stride) {
        float4 f = *(const float4*)(in + i);
        unsigned lo = (unsigned)f2bf(f.x) | ((unsigned)f2bf(f.y) << 16);
        unsigned hi = (unsigned)f2bf(f.z) | ((unsigned)f2bf(f.w) << 16);
        uint2 p; p.x = lo; p.y = hi;
        *(uint2*)(out + i) = p;
    }
}

// ---------------------------------------------------------------------------
// Fragment load from an LDS bf16 tile laid out [row][BKK] (row-major).
// 16-bit 16x32 fragment layout (ISA 7.12.2): lane m / m+16 hold row m; pairs
// carry K = {0..7,16..23} for lanes 0-15 and K = {8..15,24..31} for 16-31.
// ---------------------------------------------------------------------------
__device__ __forceinline__ bf16x16 load_frag(const unsigned short* __restrict__ tile,
                                             int rowBase, int lane) {
    int row  = rowBase + (lane & 15);
    int half = lane >> 4;
    const unsigned short* base = tile + row * BKK + 8 * half;
    uint4 lo = *(const uint4*)(base);        // K = 8*half + 0..7
    uint4 hi = *(const uint4*)(base + 16);   // K = 16 + 8*half + 0..7
    uint8v u;
    u[0] = lo.x; u[1] = lo.y; u[2] = lo.z; u[3] = lo.w;
    u[4] = hi.x; u[5] = hi.y; u[6] = hi.z; u[7] = hi.w;
    return __builtin_bit_cast(bf16x16, u);
}

// ---------------------------------------------------------------------------
// 2a) WMMA GEMM (fallback): C f32 = A_f32[M,K] * W_f32[N,K]^T, convert-on-stage
// ---------------------------------------------------------------------------
__global__ __launch_bounds__(256) void gemm_bf16_wmma(const float* __restrict__ A,
                                                      const float* __restrict__ W,
                                                      float* __restrict__ C,
                                                      int M, int N, int K) {
    __shared__ unsigned short As[BM * BKK];
    __shared__ unsigned short Bs[BN * BKK];

    const int mBlock = blockIdx.y * BM;
    const int nBlock = blockIdx.x * BN;
    const int tid    = threadIdx.x;
    const int wid    = tid >> 5;
    const int lane   = tid & 31;
    const int wr     = wid >> 1;
    const int wc     = wid & 1;

    v8f acc[2][4];
#pragma unroll
    for (int i = 0; i < 2; ++i)
#pragma unroll
        for (int j = 0; j < 4; ++j) acc[i][j] = (v8f)(0.0f);

    const int r    = tid >> 1;
    const int part = tid & 1;

    for (int k0 = 0; k0 < K; k0 += BKK) {
        __syncthreads();
        {
            const float* src = A + (size_t)(mBlock + r) * K + k0 + part * 16;
            unsigned short* dst = As + r * BKK + part * 16;
#pragma unroll
            for (int i = 0; i < 16; i += 4) {
                float4 f = *(const float4*)(src + i);
                dst[i + 0] = f2bf(f.x); dst[i + 1] = f2bf(f.y);
                dst[i + 2] = f2bf(f.z); dst[i + 3] = f2bf(f.w);
            }
        }
        {
            const float* src = W + (size_t)(nBlock + r) * K + k0 + part * 16;
            unsigned short* dst = Bs + r * BKK + part * 16;
#pragma unroll
            for (int i = 0; i < 16; i += 4) {
                float4 f = *(const float4*)(src + i);
                dst[i + 0] = f2bf(f.x); dst[i + 1] = f2bf(f.y);
                dst[i + 2] = f2bf(f.z); dst[i + 3] = f2bf(f.w);
            }
        }
        __syncthreads();

        bf16x16 afrag[2], bfrag[4];
#pragma unroll
        for (int mt = 0; mt < 2; ++mt) afrag[mt] = load_frag(As, wr * 32 + mt * 16, lane);
#pragma unroll
        for (int nt = 0; nt < 4; ++nt) bfrag[nt] = load_frag(Bs, wc * 64 + nt * 16, lane);

#pragma unroll
        for (int mt = 0; mt < 2; ++mt)
#pragma unroll
            for (int nt = 0; nt < 4; ++nt)
                acc[mt][nt] = __builtin_amdgcn_wmma_f32_16x16x32_bf16(
                    false, afrag[mt], false, bfrag[nt],
                    (short)0, acc[mt][nt], false, false);
    }

    const int halfl = lane >> 4;
    const int lan   = lane & 15;
#pragma unroll
    for (int mt = 0; mt < 2; ++mt)
#pragma unroll
        for (int nt = 0; nt < 4; ++nt) {
            int m0 = mBlock + wr * 32 + mt * 16 + halfl * 8;
            int n  = nBlock + wc * 64 + nt * 16 + lan;
#pragma unroll
            for (int rr = 0; rr < 8; ++rr)
                C[(size_t)(m0 + rr) * N + n] = acc[mt][nt][rr];
        }
}

// ---------------------------------------------------------------------------
// 2b) WMMA GEMM (preferred): bf16 inputs, double-buffered LDS tiles filled by
//     GLOBAL_LOAD_ASYNC_TO_LDS_B128 (ASYNCcnt), overlapped with WMMA compute.
// ---------------------------------------------------------------------------
__device__ __forceinline__ void async_tile(const unsigned short* __restrict__ g,
                                           int ldK, int rowBase, int k0,
                                           unsigned ldsBase, int tid) {
    // tile = 128 rows x 32 bf16 = 512 x 16B chunks; 256 threads x 2 chunks
#pragma unroll
    for (int c = tid; c < 512; c += 256) {
        int row = c >> 2, seg = c & 3;
        unsigned long long ga = (unsigned long long)(uintptr_t)(
            (const char*)g + ((size_t)(rowBase + row) * ldK + k0) * 2 + seg * 16);
        unsigned la = ldsBase + row * (BKK * 2) + seg * 16;
        asm volatile("global_load_async_to_lds_b128 %0, %1, off"
                     :: "v"(la), "v"(ga) : "memory");
    }
}

__global__ __launch_bounds__(256) void gemm_bf16_wmma_async(
        const unsigned short* __restrict__ A,   // bf16 [M,K]
        const unsigned short* __restrict__ W,   // bf16 [N,K]
        float* __restrict__ C, int M, int N, int K) {
    __shared__ unsigned short sh[4 * TILE_ELEMS];   // As[2] | Bs[2] = 32 KB
    unsigned short* Asb = sh;
    unsigned short* Bsb = sh + 2 * TILE_ELEMS;
    const unsigned shBase = (unsigned)(uintptr_t)&sh[0];   // LDS byte offset
    const unsigned asBase = shBase;
    const unsigned bsBase = shBase + 2 * TILE_BYTES;

    const int mBlock = blockIdx.y * BM;
    const int nBlock = blockIdx.x * BN;
    const int tid    = threadIdx.x;
    const int wid    = tid >> 5;
    const int lane   = tid & 31;
    const int wr     = wid >> 1;
    const int wc     = wid & 1;

    v8f acc[2][4];
#pragma unroll
    for (int i = 0; i < 2; ++i)
#pragma unroll
        for (int j = 0; j < 4; ++j) acc[i][j] = (v8f)(0.0f);

    // prologue: stage tile 0 into buffer 0
    async_tile(A, K, mBlock, 0, asBase, tid);
    async_tile(W, K, nBlock, 0, bsBase, tid);

    int buf = 0;
    for (int k0 = 0; k0 < K; k0 += BKK) {
        asm volatile("s_wait_asynccnt 0x0" ::: "memory");  // this wave's copies done
        __syncthreads();                                   // all waves' copies done
        if (k0 + BKK < K) {                                // overlap next-tile copies
            async_tile(A, K, mBlock, k0 + BKK, asBase + (buf ^ 1) * TILE_BYTES, tid);
            async_tile(W, K, nBlock, k0 + BKK, bsBase + (buf ^ 1) * TILE_BYTES, tid);
        }

        const unsigned short* At = Asb + buf * TILE_ELEMS;
        const unsigned short* Bt = Bsb + buf * TILE_ELEMS;
        bf16x16 afrag[2], bfrag[4];
#pragma unroll
        for (int mt = 0; mt < 2; ++mt) afrag[mt] = load_frag(At, wr * 32 + mt * 16, lane);
#pragma unroll
        for (int nt = 0; nt < 4; ++nt) bfrag[nt] = load_frag(Bt, wc * 64 + nt * 16, lane);

#pragma unroll
        for (int mt = 0; mt < 2; ++mt)
#pragma unroll
            for (int nt = 0; nt < 4; ++nt)
                acc[mt][nt] = __builtin_amdgcn_wmma_f32_16x16x32_bf16(
                    false, afrag[mt], false, bfrag[nt],
                    (short)0, acc[mt][nt], false, false);
        buf ^= 1;
    }

    const int halfl = lane >> 4;
    const int lan   = lane & 15;
#pragma unroll
    for (int mt = 0; mt < 2; ++mt)
#pragma unroll
        for (int nt = 0; nt < 4; ++nt) {
            int m0 = mBlock + wr * 32 + mt * 16 + halfl * 8;
            int n  = nBlock + wc * 64 + nt * 16 + lan;
#pragma unroll
            for (int rr = 0; rr < 8; ++rr)
                C[(size_t)(m0 + rr) * N + n] = acc[mt][nt][rr];
        }
}

// ---------------------------------------------------------------------------
// 3) g = -softplus(x @ Wg^T), beta = sigmoid(x @ Wbeta^T)
// ---------------------------------------------------------------------------
__global__ __launch_bounds__(256) void gb_kernel(const float* __restrict__ x,
                                                 const float* __restrict__ Wg,
                                                 const float* __restrict__ Wb,
                                                 float* __restrict__ g,
                                                 float* __restrict__ beta) {
    __shared__ float xs[HIDDEN];
    __shared__ float part[64][4];
    const int row = blockIdx.x;
    const int tid = threadIdx.x;
    const float* xr = x + (size_t)row * HIDDEN;
    for (int i = tid * 4; i < HIDDEN; i += 256 * 4) {
        float4 f = *(const float4*)(xr + i);
        xs[i + 0] = f.x; xs[i + 1] = f.y; xs[i + 2] = f.z; xs[i + 3] = f.w;
    }
    __syncthreads();

    const int d = tid & 63;
    const int c = tid >> 6;
    const int h = d & 31;
    const int which = d >> 5;
    const float* w = (which ? Wb : Wg) + (size_t)h * HIDDEN + c * 512;
    float s = 0.f;
#pragma unroll 8
    for (int i = 0; i < 512; ++i) s += xs[c * 512 + i] * w[i];
    part[d][c] = s;
    __syncthreads();

    if (tid < 64) {
        float z = part[tid][0] + part[tid][1] + part[tid][2] + part[tid][3];
        int h2 = tid & 31;
        if (tid < 32) {
            float sp = (z > 20.f) ? z : log1pf(__expf(z));
            g[(size_t)row * HV + h2] = -sp;
        } else {
            beta[(size_t)row * HV + h2] = 1.f / (1.f + __expf(-z));
        }
    }
}

// ---------------------------------------------------------------------------
// 4) In-place l2norm over 128-element segments
// ---------------------------------------------------------------------------
__global__ __launch_bounds__(128) void norm_kernel(float* __restrict__ data,
                                                   float extraScale) {
    __shared__ float red[128];
    const size_t base = (size_t)blockIdx.x * 128;
    const int t = threadIdx.x;
    float v = data[base + t];
    red[t] = v * v;
    __syncthreads();
    for (int s = 64; s > 0; s >>= 1) {
        if (t < s) red[t] += red[t + s];
        __syncthreads();
    }
    float scale = rsqrtf(red[0] + 1e-6f) * extraScale;
    data[base + t] = v * scale;
}

// ---------------------------------------------------------------------------
// 5) Gated delta rule: one block per (b, hv-head); 128x128 f32 state in LDS.
//    OutT = unsigned short (bf16, async path) or float (fallback).
// ---------------------------------------------------------------------------
template <typename OutT>
__global__ __launch_bounds__(128) void delta_kernel(const float* __restrict__ qn,
                                                    const float* __restrict__ kn,
                                                    const float* __restrict__ v,
                                                    const float* __restrict__ g,
                                                    const float* __restrict__ beta,
                                                    OutT* __restrict__ out) {
    extern __shared__ float smem[];
    float* Sst  = smem;                 // 128*128
    float* kbuf = Sst + DK * DV;        // 128
    float* qbuf = kbuf + DK;            // 128
    float* vbuf = qbuf + DK;            // 128
    float* sc   = vbuf + DV;            // 2

    const int bh = blockIdx.x;
    const int b  = bh >> 5;
    const int h  = bh & 31;
    const int hk = h >> 1;
    const int tv = threadIdx.x;

    for (int i = tv; i < DK * DV; i += 128) Sst[i] = 0.f;

    for (int t = 0; t < SS; ++t) {
        __syncthreads();
        const size_t row = (size_t)(b * SS + t);
        kbuf[tv] = kn[row * (HK * DK) + hk * DK + tv];
        qbuf[tv] = qn[row * (HK * DK) + hk * DK + tv];
        vbuf[tv] = v [row * (HV * DV) + h  * DV + tv];
        if (tv == 0) {
            sc[0] = __expf(g[row * HV + h]);
            sc[1] = beta[row * HV + h];
        }
        if (t + 8 < SS) {   // hide fetch latency for future steps
            const size_t rp = (size_t)(b * SS + t + 8);
            __builtin_prefetch(&kn[rp * (HK * DK) + hk * DK + tv], 0, 0);
            __builtin_prefetch(&qn[rp * (HK * DK) + hk * DK + tv], 0, 0);
            __builtin_prefetch(&v [rp * (HV * DV) + h  * DV + tv], 0, 0);
        }
        __syncthreads();
        const float eg = sc[0], bb = sc[1];

        float tmp = 0.f;
#pragma unroll 8
        for (int kk = 0; kk < DK; ++kk) tmp += Sst[kk * DV + tv] * kbuf[kk];
        const float verr = (vbuf[tv] - eg * tmp) * bb;

        float o = 0.f;
#pragma unroll 8
        for (int kk = 0; kk < DK; ++kk) {
            float sn = Sst[kk * DV + tv] * eg + kbuf[kk] * verr;
            Sst[kk * DV + tv] = sn;
            o += sn * qbuf[kk];
        }
        if constexpr (sizeof(OutT) == 2)
            out[row * (HV * DV) + h * DV + tv] = (OutT)f2bf(o);
        else
            out[row * (HV * DV) + h * DV + tv] = (OutT)o;
    }
}

// ---------------------------------------------------------------------------
// Launch
// ---------------------------------------------------------------------------
extern "C" void kernel_launch(void* const* d_in, const int* in_sizes, int n_in,
                              void* d_out, int out_size, void* d_ws, size_t ws_size,
                              hipStream_t stream) {
    (void)in_sizes; (void)n_in; (void)out_size;

    const int*   ids   = (const int*)d_in[0];
    const float* emb   = (const float*)d_in[1];
    const float* Wq    = (const float*)d_in[2];
    const float* Wk    = (const float*)d_in[3];
    const float* Wv    = (const float*)d_in[4];
    const float* Wg    = (const float*)d_in[5];
    const float* Wbeta = (const float*)d_in[6];
    const float* Wo    = (const float*)d_in[7];
    float* out = (float*)d_out;

    const size_t szX   = (size_t)MROWS * HIDDEN * 4;
    const size_t szQK  = (size_t)MROWS * (HK * DK) * 4;
    const size_t szV   = (size_t)MROWS * (HV * DV) * 4;
    const size_t szGB  = (size_t)MROWS * HV * 4;
    const size_t szXb  = (size_t)MROWS * HIDDEN * 2;
    const size_t szODb = (size_t)MROWS * (HV * DV) * 2;
    const size_t szWqb = (size_t)(HK * DK) * HIDDEN * 2;
    const size_t szWvb = (size_t)(HV * DV) * HIDDEN * 2;
    const size_t szWob = (size_t)HIDDEN * (HV * DV) * 2;

    const size_t needAsync =
        szX + szXb + 2 * szQK + szV + 2 * szGB + szODb + 2 * szWqb + szWvb + szWob;
    const bool useAsync = (ws_size >= needAsync);

    char* ws = (char*)d_ws;
    float* x  = (float*)ws;  ws += szX;
    float* q  = (float*)ws;  ws += szQK;
    float* k  = (float*)ws;  ws += szQK;
    float* v  = (float*)ws;  ws += szV;
    float* g  = (float*)ws;  ws += szGB;
    float* bt = (float*)ws;  ws += szGB;

    const size_t dynLds = (size_t)(DK * DV + 3 * DK + 2) * sizeof(float);

    if (useAsync) {
        unsigned short* xb  = (unsigned short*)ws;  ws += szXb;
        unsigned short* odb = (unsigned short*)ws;  ws += szODb;
        unsigned short* Wqb = (unsigned short*)ws;  ws += szWqb;
        unsigned short* Wkb = (unsigned short*)ws;  ws += szWqb;
        unsigned short* Wvb = (unsigned short*)ws;  ws += szWvb;
        unsigned short* Wob = (unsigned short*)ws;  ws += szWob;

        embed_kernel<<<MROWS, 256, 0, stream>>>(ids, emb, x, xb);
        cast_f32_bf16<<<1024, 256, 0, stream>>>(Wq, Wqb, (long long)(HK * DK) * HIDDEN);
        cast_f32_bf16<<<1024, 256, 0, stream>>>(Wk, Wkb, (long long)(HK * DK) * HIDDEN);
        cast_f32_bf16<<<1024, 256, 0, stream>>>(Wv, Wvb, (long long)(HV * DV) * HIDDEN);
        cast_f32_bf16<<<1024, 256, 0, stream>>>(Wo, Wob, (long long)HIDDEN * (HV * DV));

        gemm_bf16_wmma_async<<<dim3((HK * DK) / BN, MROWS / BM), 256, 0, stream>>>(
            xb, Wqb, q, MROWS, HK * DK, HIDDEN);
        gemm_bf16_wmma_async<<<dim3((HK * DK) / BN, MROWS / BM), 256, 0, stream>>>(
            xb, Wkb, k, MROWS, HK * DK, HIDDEN);
        gemm_bf16_wmma_async<<<dim3((HV * DV) / BN, MROWS / BM), 256, 0, stream>>>(
            xb, Wvb, v, MROWS, HV * DV, HIDDEN);

        gb_kernel<<<MROWS, 256, 0, stream>>>(x, Wg, Wbeta, g, bt);
        norm_kernel<<<MROWS * HK, 128, 0, stream>>>(q, 0.08838834764831845f);
        norm_kernel<<<MROWS * HK, 128, 0, stream>>>(k, 1.0f);

        delta_kernel<unsigned short><<<BB * HV, 128, dynLds, stream>>>(
            q, k, v, g, bt, odb);

        gemm_bf16_wmma_async<<<dim3(HIDDEN / BN, MROWS / BM), 256, 0, stream>>>(
            odb, Wob, out, MROWS, HIDDEN, HV * DV);
    } else {
        float* od = (float*)ws;  ws += szV;

        embed_kernel<<<MROWS, 256, 0, stream>>>(ids, emb, x, (unsigned short*)nullptr);

        gemm_bf16_wmma<<<dim3((HK * DK) / BN, MROWS / BM), 256, 0, stream>>>(
            x, Wq, q, MROWS, HK * DK, HIDDEN);
        gemm_bf16_wmma<<<dim3((HK * DK) / BN, MROWS / BM), 256, 0, stream>>>(
            x, Wk, k, MROWS, HK * DK, HIDDEN);
        gemm_bf16_wmma<<<dim3((HV * DV) / BN, MROWS / BM), 256, 0, stream>>>(
            x, Wv, v, MROWS, HV * DV, HIDDEN);

        gb_kernel<<<MROWS, 256, 0, stream>>>(x, Wg, Wbeta, g, bt);
        norm_kernel<<<MROWS * HK, 128, 0, stream>>>(q, 0.08838834764831845f);
        norm_kernel<<<MROWS * HK, 128, 0, stream>>>(k, 1.0f);

        delta_kernel<float><<<BB * HV, 128, dynLds, stream>>>(q, k, v, g, bt, od);

        gemm_bf16_wmma<<<dim3(HIDDEN / BN, MROWS / BM), 256, 0, stream>>>(
            od, Wo, out, MROWS, HIDDEN, HV * DV);
    }
}
